// AttnScenePooling_45286135169029
// MI455X (gfx1250) — compile-verified, matrix operand order checked by reading, not run on hardware
//
#include <hip/hip_runtime.h>
#include <hip/hip_bf16.h>
#include <math.h>

#define IN_DIM 512
#define HID    256
#define PTS    32          // points per score block
#define POOL_CHUNKS 8

typedef __attribute__((ext_vector_type(16))) _Float16 v16h;
typedef __attribute__((ext_vector_type(8)))  float    v8f;

// ---------------- w1 fp32 -> f16 (runs once per launch, w1 = 512 KB, L2-resident) ----
__global__ void cvt_w1_f16(const float* __restrict__ w1, _Float16* __restrict__ w1h, int n) {
  int i = blockIdx.x * blockDim.x + threadIdx.x;
  if (i < n) w1h[i] = (_Float16)w1[i];
}

__global__ void zero_f32(float* __restrict__ p, int n) {
  int i = blockIdx.x * blockDim.x + threadIdx.x;
  if (i < n) p[i] = 0.f;
}

// ---------------- fused LayerNorm -> WMMA GEMM -> GELU -> w2 dot => score ------------
__global__ __launch_bounds__(256) void score_kernel(
    const float* __restrict__ feats, const float* __restrict__ ln_g,
    const float* __restrict__ ln_b,  const _Float16* __restrict__ w1h,
    const float* __restrict__ b1,    const float* __restrict__ w2,
    const float* __restrict__ b2,    float* __restrict__ s_out, int npts)
{
  __shared__ _Float16 xln[PTS][IN_DIM];   // 32 KB f16 LN output tile
  __shared__ float    s_part[PTS];

  const int tid  = threadIdx.x;
  const int wave = tid >> 5;
  const int lane = tid & 31;

  if (tid < PTS) s_part[tid] = 0.f;

  // ---- LayerNorm: each of 8 waves normalizes 4 points (512 feats via 32 lanes) ----
  for (int q = 0; q < 4; ++q) {
    const int pl = wave * 4 + q;
    const int p  = blockIdx.x * PTS + pl;
    if (p < npts) {
      float f[16];
      float sum = 0.f, ssq = 0.f;
      #pragma unroll
      for (int j = 0; j < 16; ++j) {
        f[j] = feats[(size_t)p * IN_DIM + lane + 32 * j];
        sum += f[j];
        ssq += f[j] * f[j];
      }
      #pragma unroll
      for (int o = 16; o > 0; o >>= 1) {
        sum += __shfl_xor(sum, o, 32);
        ssq += __shfl_xor(ssq, o, 32);
      }
      const float mu  = sum * (1.f / IN_DIM);
      const float var = ssq * (1.f / IN_DIM) - mu * mu;
      const float rs  = rsqrtf(var + 1e-5f);
      #pragma unroll
      for (int j = 0; j < 16; ++j) {
        const int c = lane + 32 * j;
        const float v = (f[j] - mu) * rs * ln_g[c] + ln_b[c];
        xln[pl][c] = (_Float16)v;
      }
    } else {
      #pragma unroll
      for (int j = 0; j < 16; ++j) xln[pl][lane + 32 * j] = (_Float16)0.f;
    }
  }
  __syncthreads();

  // ---- WMMA: h = xln[32x512] @ w1^T[512x256]; 2x16 grid of 16x16 tiles, 8 waves ----
  const int hi = lane >> 4;    // half-wave selects K sub-block per ISA layout
  const int lm = lane & 15;
  const unsigned int* __restrict__ w1u = (const unsigned int*)w1h;

  for (int t = 0; t < 4; ++t) {
    const int tile   = wave + t * 8;       // 0..31
    const int m_base = (tile >> 4) * 16;   // 2 row tiles
    const int n_base = (tile & 15) * 16;   // 16 col tiles
    const int n      = n_base + lm;        // this lane's output column (hidden idx)

    v8f acc = {};
    for (int k0 = 0; k0 < IN_DIM; k0 += 32) {
      union { v16h v; unsigned int u[8]; } A, B;
      #pragma unroll
      for (int j = 0; j < 8; ++j) {
        // ISA 16-bit 16x32 A layout: VGPR j<4 -> K = k0+2j(+8 for hi lanes),
        //                            VGPR j>=4 -> K = k0+16+2(j-4)(+8 for hi lanes)
        const int kk = k0 + ((j < 4) ? (2 * j) : (16 + 2 * (j - 4))) + hi * 8;
        A.u[j] = *(const unsigned int*)&xln[m_base + lm][kk];      // ds_load_b32
        B.u[j] = w1u[n * (IN_DIM / 2) + (kk >> 1)];                // column-major B per lane
      }
      acc = __builtin_amdgcn_wmma_f32_16x16x32_f16(
          false, A.v, false, B.v, (short)0, acc, false, false);
    }

    // Epilogue: +b1, exact GELU, *w2, reduce across N into per-point score
    const float b1n = b1[n];
    const float w2n = w2[n];
    #pragma unroll
    for (int r = 0; r < 8; ++r) {
      const int   m = m_base + r + 8 * hi;   // D layout: row = r + 8*hi, col = lane&15
      const float h = acc[r] + b1n;
      const float g = 0.5f * h * (1.f + erff(h * 0.70710678118f));
      atomicAdd(&s_part[m], g * w2n);        // ds_add_f32
    }
  }
  __syncthreads();

  if (tid < PTS) {
    const int p = blockIdx.x * PTS + tid;
    if (p < npts) s_out[p] = s_part[tid] + b2[0];
  }
}

// ---------------- per-scene softmax stats (max, sum-exp) ----------------------------
__global__ __launch_bounds__(256) void stats_kernel(
    const float* __restrict__ s, const int* __restrict__ offsets,
    float* __restrict__ smax, float* __restrict__ denom)
{
  const int seg = blockIdx.x;
  const int beg = offsets[seg], end = offsets[seg + 1];
  const int tid = threadIdx.x, wv = tid >> 5, ln = tid & 31;
  __shared__ float red[8];

  float m = -3.402823466e38f;
  for (int i = beg + tid; i < end; i += blockDim.x) m = fmaxf(m, s[i]);
  #pragma unroll
  for (int o = 16; o > 0; o >>= 1) m = fmaxf(m, __shfl_xor(m, o, 32));
  if (ln == 0) red[wv] = m;
  __syncthreads();
  if (tid == 0) {
    float mm = red[0];
    for (int i = 1; i < 8; ++i) mm = fmaxf(mm, red[i]);
    red[0] = mm;
  }
  __syncthreads();
  m = red[0];
  __syncthreads();

  float sum = 0.f;
  for (int i = beg + tid; i < end; i += blockDim.x) sum += expf(s[i] - m);
  #pragma unroll
  for (int o = 16; o > 0; o >>= 1) sum += __shfl_xor(sum, o, 32);
  if (ln == 0) red[wv] = sum;
  __syncthreads();
  if (tid == 0) {
    float tt = 0.f;
    for (int i = 0; i < 8; ++i) tt += red[i];
    smax[seg]  = m;
    denom[seg] = tt;
  }
}

// ---------------- weighted pooling: out[seg,d] = sum_p w_p * feats[p,d] -------------
__global__ __launch_bounds__(512) void pool_kernel(
    const float* __restrict__ feats, const float* __restrict__ s,
    const int* __restrict__ offsets, const float* __restrict__ smax,
    const float* __restrict__ denom, float* __restrict__ out)
{
  const int seg = blockIdx.x / POOL_CHUNKS;
  const int ch  = blockIdx.x % POOL_CHUNKS;
  const int beg = offsets[seg], end = offsets[seg + 1];
  const int len = end - beg;
  const int per = (len + POOL_CHUNKS - 1) / POOL_CHUNKS;
  const int cbeg = beg + ch * per;
  const int cend = (cbeg + per < end) ? (cbeg + per) : end;
  if (cbeg >= cend) return;   // uniform per block

  __shared__ float wts[512];
  const float sm     = smax[seg];
  const float inv_dn = 1.f / denom[seg];
  const int   d      = threadIdx.x;   // one feature dim per thread (512 = IN_DIM)
  float acc = 0.f;

  for (int p0 = cbeg; p0 < cend; p0 += 512) {
    const int slab = (cend - p0 < 512) ? (cend - p0) : 512;
    __syncthreads();
    if (d < slab) wts[d] = expf(s[p0 + d] - sm) * inv_dn;
    __syncthreads();
    for (int q = 0; q < slab; ++q)
      acc += feats[(size_t)(p0 + q) * IN_DIM + d] * wts[q];
  }
  atomicAdd(&out[seg * IN_DIM + d], acc);   // global_atomic_add_f32
}

// ---------------- launcher ----------------------------------------------------------
extern "C" void kernel_launch(void* const* d_in, const int* in_sizes, int n_in,
                              void* d_out, int out_size, void* d_ws, size_t ws_size,
                              hipStream_t stream)
{
  (void)n_in; (void)ws_size;
  const float* feats = (const float*)d_in[0];
  const float* ln_g  = (const float*)d_in[1];
  const float* ln_b  = (const float*)d_in[2];
  const float* w1    = (const float*)d_in[3];
  const float* b1    = (const float*)d_in[4];
  const float* w2    = (const float*)d_in[5];
  const float* b2    = (const float*)d_in[6];
  const int* offsets = (const int*)d_in[7];

  const int npts = in_sizes[0] / IN_DIM;
  const int nseg = in_sizes[7] - 1;
  float* out = (float*)d_out;

  char* ws = (char*)d_ws;
  _Float16* w1h  = (_Float16*)ws;                                  // 256 KB
  float*    s    = (float*)(ws + (size_t)HID * IN_DIM * sizeof(_Float16));
  float*    smax = (float*)((char*)s + (size_t)npts * sizeof(float));
  float*    denm = smax + nseg;

  cvt_w1_f16<<<(HID * IN_DIM + 255) / 256, 256, 0, stream>>>(w1, w1h, HID * IN_DIM);
  zero_f32<<<(out_size + 255) / 256, 256, 0, stream>>>(out, out_size);
  score_kernel<<<(npts + PTS - 1) / PTS, 256, 0, stream>>>(
      feats, ln_g, ln_b, w1h, b1, w2, b2, s, npts);
  stats_kernel<<<nseg, 256, 0, stream>>>(s, offsets, smax, denm);
  pool_kernel<<<nseg * POOL_CHUNKS, 512, 0, stream>>>(feats, s, offsets, smax, denm, out);
}
